// LLModel_31610959299083
// MI455X (gfx1250) — matrix operand order repaired and checked by reading, not compiled, not run on hardware
//
#include <hip/hip_runtime.h>
#include <hip/hip_bf16.h>

// CDNA5 (gfx1250) wave32 WMMA types
typedef __attribute__((ext_vector_type(16))) __bf16 v16bf;
typedef __attribute__((ext_vector_type(8)))  float  v8f;

#define DEV static __device__ __forceinline__

// ---------------- bf16 conversion helpers (round-half-up: 1 VALU op) -------
DEV unsigned short f2bf(float f) {
  unsigned u = __builtin_bit_cast(unsigned, f);
  return (unsigned short)((u + 0x8000u) >> 16);   // enables *_d16_hi stores
}
DEV __bf16 bfbits(unsigned short h) { return __builtin_bit_cast(__bf16, h); }

#if defined(__AMDGCN__) && __has_builtin(__builtin_amdgcn_cvt_pk_bf16_f32)
DEV unsigned pk2bf(float lo, float hi) {          // v_cvt_pk_bf16_f32
  typedef __attribute__((ext_vector_type(2))) __bf16 v2bf;
  v2bf v = __builtin_amdgcn_cvt_pk_bf16_f32(lo, hi);
  return __builtin_bit_cast(unsigned, v);
}
#elif defined(__AMDGCN__) && __has_builtin(__builtin_amdgcn_perm)
DEV unsigned pk2bf(float lo, float hi) {          // 2x v_add + v_perm_b32
  unsigned ul = __builtin_bit_cast(unsigned, lo) + 0x8000u;
  unsigned uh = __builtin_bit_cast(unsigned, hi) + 0x8000u;
  return __builtin_amdgcn_perm(uh, ul, 0x07060302u);  // {uh.hi16, ul.hi16}
}
#else
DEV unsigned pk2bf(float lo, float hi) {
  return (unsigned)f2bf(lo) | ((unsigned)f2bf(hi) << 16);
}
#endif

// ---------------- async global->LDS (ASYNCcnt path), guarded ---------------
#if defined(__AMDGCN__) && __has_builtin(__builtin_amdgcn_global_load_async_to_lds_b128)
#define HAVE_ASYNC_LDS 1
typedef int v4i_vs __attribute__((vector_size(4 * sizeof(int))));
typedef __attribute__((address_space(1))) v4i_vs glb4_t;   // global (AS1)
typedef __attribute__((address_space(3))) v4i_vs lds4_t;   // LDS (AS3)
DEV glb4_t* gptr4(const void* p) {
  return (glb4_t*)(unsigned long long)p;                    // full flat address
}
DEV lds4_t* lptr4(const void* p) {
  return (lds4_t*)(unsigned)(unsigned long long)p;          // low 32b = LDS offset
}
template <int OFF>
DEV void async_copy_b128(const unsigned short* gsrc, const unsigned short* ldst) {
  __builtin_amdgcn_global_load_async_to_lds_b128(gptr4(gsrc), lptr4(ldst), OFF, 0);
}
DEV void wait_async0() {
#if __has_builtin(__builtin_amdgcn_s_wait_asynccnt)
  __builtin_amdgcn_s_wait_asynccnt(0);
#else
  asm volatile("s_wait_asynccnt 0x0" ::: "memory");
#endif
}
#endif

// ---------------- WMMA fragment builders (contiguous K order in A and B) --------------
// A (16x32 bf16): lanes 0-15 row M=l hold K{0..15}; lanes 16-31 K{16..31}
DEV v16bf make_afrag(const unsigned short* rowp, int hi) {
  v16bf a;
#pragma unroll
  for (int vg = 0; vg < 8; ++vg) {
    const int kb = 2 * vg + hi * 16;
    a[2 * vg]     = bfbits(rowp[kb]);
    a[2 * vg + 1] = bfbits(rowp[kb + 1]);
  }
  return a;
}
// B (32x16 bf16): lane l -> column N=l; element j -> K = 16*hi + j
// (caller pre-offsets pointer by hi*16) -> 16 contiguous bf16 per lane.
DEV v16bf make_bfrag(const unsigned short* p) {
  v16bf b;
#pragma unroll
  for (int j = 0; j < 16; ++j) b[j] = bfbits(p[j]);
  return b;
}
DEV v8f wmma_bf16(v16bf a, v16bf b, v8f c) {
  return __builtin_amdgcn_wmma_f32_16x16x32_bf16(false, a, false, b,
                                                 (short)0, c, false, false);
}
// Row reductions over the 16 lanes holding one C-row (masks stay inside halves)
DEV float redmax16(float x) {
#pragma unroll
  for (int m = 1; m <= 8; m <<= 1) x = fmaxf(x, __shfl_xor(x, m, 32));
  return x;
}
DEV float redsum16(float x) {
#pragma unroll
  for (int m = 1; m <= 8; m <<= 1) x += __shfl_xor(x, m, 32);
  return x;
}

// ---------------------------------------------------------------------------
// Kernel 1: fused QKV projection (bf16 WMMA, f32 accum) + RoPE epilogue.
// grid (4096/256, 1024/64, 3), block 256 (8 waves). Block tile 256x64, K step 32.
// Each wave owns TWO 16-row M-tiles -> every B-fragment feeds 2 WMMAs.
// Outputs bf16 Q/K/V in [B,H,S,64] layout for the attention kernel.
// ---------------------------------------------------------------------------
__global__ __launch_bounds__(256)
void k_qkv_rope(const float* __restrict__ X,  const float* __restrict__ Wq,
                const float* __restrict__ Wk, const float* __restrict__ Wv,
                const int*   __restrict__ pos,
                unsigned short* __restrict__ Qb,
                unsigned short* __restrict__ Kb,
                unsigned short* __restrict__ Vb) {
  constexpr int D = 1024, S = 2048;
  constexpr int XS = 40, WS = 40;                 // 80B rows: 16B aligned, 20 banks
  __shared__ unsigned short Xs[256 * XS];
  __shared__ unsigned short Ws[64 * WS];

  const int m0 = blockIdx.x * 256;
  const int n0 = blockIdx.y * 64;
  const int z  = blockIdx.z;                      // 0=Q 1=K 2=V
  const float* W = (z == 0) ? Wq : (z == 1) ? Wk : Wv;

  const int t = threadIdx.x, lane = t & 31, w = t >> 5;
  const int l = lane & 15, hi = lane >> 4;

  v8f acc[2][4] = {};

  for (int k0 = 0; k0 < D; k0 += 32) {
    __syncthreads();
    { // stage X tile 256x32 (f32 -> bf16): one full row per thread
      const float4* src =
          reinterpret_cast<const float4*>(X + (size_t)(m0 + t) * D + k0);
      float4 f[8];
#pragma unroll
      for (int j = 0; j < 8; ++j) f[j] = src[j];
      unsigned* dst = reinterpret_cast<unsigned*>(Xs + t * XS);
#pragma unroll
      for (int j = 0; j < 8; ++j) {
        dst[2 * j]     = pk2bf(f[j].x, f[j].y);
        dst[2 * j + 1] = pk2bf(f[j].z, f[j].w);
      }
    }
    { // stage W tile 64x32; B[k][n] = W[n][k] (row-major is B-ready)
      const int row = t >> 2, cb = (t & 3) * 8;
      const float4* src =
          reinterpret_cast<const float4*>(W + (size_t)(n0 + row) * D + k0 + cb);
      float4 f0 = src[0], f1 = src[1];
      unsigned* dst = reinterpret_cast<unsigned*>(Ws + row * WS + cb);
      dst[0] = pk2bf(f0.x, f0.y); dst[1] = pk2bf(f0.z, f0.w);
      dst[2] = pk2bf(f1.x, f1.y); dst[3] = pk2bf(f1.z, f1.w);
    }
    if (k0 + 32 < D) {   // global_prefetch_b8 of the next K tile
      __builtin_prefetch(X + (size_t)(m0 + t) * D + k0 + 32, 0, 1);
      __builtin_prefetch(W + (size_t)(n0 + (t >> 2)) * D + k0 + 32 + (t & 3) * 8, 0, 1);
    }
    __syncthreads();

    const v16bf a0 = make_afrag(Xs + (w * 32 + l) * XS, hi);
    const v16bf a1 = make_afrag(Xs + (w * 32 + 16 + l) * XS, hi);
#pragma unroll
    for (int nt = 0; nt < 4; ++nt) {
      const v16bf b = make_bfrag(Ws + (nt * 16 + l) * WS + hi * 16);
      acc[0][nt] = wmma_bf16(a0, b, acc[0][nt]);   // B reused back-to-back
      acc[1][nt] = wmma_bf16(a1, b, acc[1][nt]);
    }
  }

  // Epilogue: C layout lane l -> column n; xor-1 partner is the paired RoPE column
#pragma unroll
  for (int mt = 0; mt < 2; ++mt)
#pragma unroll
    for (int nt = 0; nt < 4; ++nt) {
      const int n = n0 + nt * 16 + l;
      const int h = n >> 6, d = n & 63;
#pragma unroll
      for (int r = 0; r < 8; ++r) {
        const int m = m0 + w * 32 + mt * 16 + r + hi * 8;
        const int b = m >> 11, s = m & 2047;
        float x = acc[mt][nt][r];
        const float xp = __shfl_xor(x, 1, 32);         // partner column value
        const size_t idx = ((size_t)(b * 16 + h) * S + s) * 64 + d;
        if (z == 2) {
          Vb[idx] = f2bf(x);
        } else {
          const int i = d >> 1;
          const float ang = (float)pos[s] * __expf(-(float)i * 0.28782313663087705f);
          float sn, cs; __sincosf(ang, &sn, &cs);
          const float res = (d & 1) ? (xp * sn + x * cs)  // odd:  xe*sin + xo*cos
                                    : (x * cs - xp * sn); // even: xe*cos - xo*sin
          ((z == 0) ? Qb : Kb)[idx] = f2bf(res);
        }
      }
    }
}

// ---------------------------------------------------------------------------
// Kernel 2: causal flash attention per (b,h). grid (S/64, B*H), block 128
// (4 waves; wave w owns 16 query rows). K tile staged via async global->LDS
// (ASYNCcnt); V staged transposed so P*V B-frags are contiguous.
// ---------------------------------------------------------------------------
__global__ __launch_bounds__(128)
void k_attn(const unsigned short* __restrict__ Qb,
            const unsigned short* __restrict__ Kb,
            const unsigned short* __restrict__ Vb,
            unsigned short* __restrict__ Ctx) {
  constexpr int S = 2048, DK = 64, LS = 72;     // 144B row stride, 16B aligned
  __shared__ unsigned short Ks[64 * LS];        // [key][d]
  __shared__ unsigned short Vt[64 * LS];        // [d][key] (transposed)
  __shared__ unsigned short Ps[64 * LS];        // probs, per-wave 16-row slices

  const int qt = blockIdx.x, bh = blockIdx.y;
  const unsigned short* Q = Qb + (size_t)bh * S * DK;
  const unsigned short* K = Kb + (size_t)bh * S * DK;
  const unsigned short* V = Vb + (size_t)bh * S * DK;

  const int t = threadIdx.x, lane = t & 31, w = t >> 5;
  const int l = lane & 15, hi = lane >> 4;
  const int q0 = qt * 64;

  // Q fragments for this wave's 16 rows, kept resident (dk=64 -> 2 K-steps)
  const unsigned short* qrow = Q + (size_t)(q0 + w * 16 + l) * DK;
  const v16bf aq0 = make_afrag(qrow, hi);
  const v16bf aq1 = make_afrag(qrow + 32, hi);

  v8f ctx[4] = {};
  float m_r[8], l_r[8];
#pragma unroll
  for (int r = 0; r < 8; ++r) { m_r[r] = -1e30f; l_r[r] = 0.f; }

  for (int kt = 0; kt <= qt; ++kt) {
    __syncthreads();
    { // cooperative stage: K via async b128 copies, V via transposed scatter
      const int row = t >> 1, cb = (t & 1) * 32;
      const unsigned short* ksrc = K + (size_t)(kt * 64 + row) * DK + cb;
      unsigned short* kdst = Ks + row * LS + cb;   // 16B aligned
#if defined(HAVE_ASYNC_LDS)
      async_copy_b128<0>(ksrc, kdst);
      async_copy_b128<16>(ksrc, kdst);
      async_copy_b128<32>(ksrc, kdst);
      async_copy_b128<48>(ksrc, kdst);
#else
      const uint4* ks4 = reinterpret_cast<const uint4*>(ksrc);
      uint4* kd4 = reinterpret_cast<uint4*>(kdst);
      kd4[0] = ks4[0]; kd4[1] = ks4[1]; kd4[2] = ks4[2]; kd4[3] = ks4[3];
#endif
      const unsigned short* vsrc = V + (size_t)(kt * 64 + row) * DK + cb;
#pragma unroll
      for (int j = 0; j < 32; ++j) Vt[(cb + j) * LS + row] = vsrc[j];
      if (kt + 1 <= qt)
        __builtin_prefetch(K + (size_t)(kt * 64 + 64 + row) * DK + cb, 0, 1);
#if defined(HAVE_ASYNC_LDS)
      wait_async0();                 // K tile resident before the barrier
#endif
    }
    __syncthreads();

    // scores: S = Q * K^T  (GEMM-K = dk; 2 K-steps, 4 n-tiles)
    v8f sc[4] = {};
#pragma unroll
    for (int ks = 0; ks < 2; ++ks) {
      v16bf bk[4];
#pragma unroll
      for (int nt = 0; nt < 4; ++nt)
        bk[nt] = make_bfrag(Ks + (nt * 16 + l) * LS + ks * 32 + hi * 16);
      const v16bf aq = ks ? aq1 : aq0;
#pragma unroll
      for (int nt = 0; nt < 4; ++nt)
        sc[nt] = wmma_bf16(aq, bk[nt], sc[nt]);
    }

    // scale, causal mask, online softmax (row = r+8*hi across 16 lanes)
    float alpha[8];
#pragma unroll
    for (int r = 0; r < 8; ++r) {
      const int qg = q0 + w * 16 + r + hi * 8;
      float xs[4]; float mx = -1e30f;
#pragma unroll
      for (int nt = 0; nt < 4; ++nt) {
        const int kg = kt * 64 + nt * 16 + l;
        float x = sc[nt][r] * 0.125f;                 // 1/sqrt(64)
        x = (kg > qg) ? -1e30f : x;
        xs[nt] = x; mx = fmaxf(mx, x);
      }
      mx = redmax16(mx);
      const float mnew = fmaxf(m_r[r], mx);
      const float al = __expf(m_r[r] - mnew);
      float rs = 0.f;
#pragma unroll
      for (int nt = 0; nt < 4; ++nt) {
        const float pv = __expf(xs[nt] - mnew);
        sc[nt][r] = pv; rs += pv;
      }
      rs = redsum16(rs);
      l_r[r] = l_r[r] * al + rs;
      m_r[r] = mnew;
      alpha[r] = al;
    }
#pragma unroll
    for (int nt = 0; nt < 4; ++nt)
#pragma unroll
      for (int r = 0; r < 8; ++r) ctx[nt][r] *= alpha[r];

    // stash P (bf16) into this wave's private LDS slice (per-wave LDS in-order)
#pragma unroll
    for (int nt = 0; nt < 4; ++nt)
#pragma unroll
      for (int r = 0; r < 8; ++r)
        Ps[(w * 16 + r + hi * 8) * LS + nt * 16 + l] = f2bf(sc[nt][r]);
    asm volatile("s_wait_dscnt 0" ::: "memory");

    // ctx += P * V  (GEMM-K = 64 keys; batched frag loads then WMMA chain)
    const unsigned short* prow = Ps + (size_t)(w * 16 + l) * LS;
#pragma unroll
    for (int ks = 0; ks < 2; ++ks) {
      v16bf bv[4];
#pragma unroll
      for (int nt = 0; nt < 4; ++nt)
        bv[nt] = make_bfrag(Vt + (nt * 16 + l) * LS + ks * 32 + hi * 16);
      const v16bf ap = make_afrag(prow + ks * 32, hi);
#pragma unroll
      for (int nt = 0; nt < 4; ++nt)
        ctx[nt] = wmma_bf16(ap, bv[nt], ctx[nt]);
    }
  }

  // normalize and emit ctx as bf16 in [B,S,H*64] (GEMM-ready rows)
  const int b = bh >> 4, h = bh & 15;
  float inv[8];
#pragma unroll
  for (int r = 0; r < 8; ++r) inv[r] = (l_r[r] > 0.f) ? 1.f / l_r[r] : 0.f;
#pragma unroll
  for (int nt = 0; nt < 4; ++nt)
#pragma unroll
    for (int r = 0; r < 8; ++r) {
      const int qg = q0 + w * 16 + r + hi * 8;
      Ctx[((size_t)(b * S + qg)) * 1024 + h * 64 + nt * 16 + l] =
          f2bf(ctx[nt][r] * inv[r]);
    }
}

// ---------------------------------------------------------------------------
// Kernel 3: output projection out = ctx @ Wo^T (bf16 WMMA, f32 out).
// grid (16,16), block 256, tile 256x64, K step 32; two M-tiles per wave so
// each B-fragment feeds 2 WMMAs. ctx tile staged via async global->LDS.
// ---------------------------------------------------------------------------
__global__ __launch_bounds__(256)
void k_oproj(const unsigned short* __restrict__ Ctx,
             const float* __restrict__ Wo,
             float* __restrict__ Out) {
  constexpr int D = 1024, XS = 40, WS = 40;
  __shared__ unsigned short Xs[256 * XS];
  __shared__ unsigned short Ws[64 * WS];
  const int m0 = blockIdx.x * 256, n0 = blockIdx.y * 64;
  const int t = threadIdx.x, lane = t & 31, w = t >> 5;
  const int l = lane & 15, hi = lane >> 4;

  v8f acc[2][4] = {};
  for (int k0 = 0; k0 < D; k0 += 32) {
    __syncthreads();
    { // ctx already bf16: one 64B row chunk per thread, async global->LDS
      const unsigned short* src = Ctx + (size_t)(m0 + t) * D + k0;
      unsigned short* dst = Xs + t * XS;           // 80B rows: 16B aligned
#if defined(HAVE_ASYNC_LDS)
      async_copy_b128<0>(src, dst);
      async_copy_b128<16>(src, dst);
      async_copy_b128<32>(src, dst);
      async_copy_b128<48>(src, dst);
#else
      const uint4* s4 = reinterpret_cast<const uint4*>(src);
      uint4* d4 = reinterpret_cast<uint4*>(dst);
      d4[0] = s4[0]; d4[1] = s4[1]; d4[2] = s4[2]; d4[3] = s4[3];
#endif
    }
    {
      const int row = t >> 2, cb = (t & 3) * 8;
      const float4* src =
          reinterpret_cast<const float4*>(Wo + (size_t)(n0 + row) * D + k0 + cb);
      float4 f0 = src[0], f1 = src[1];
      unsigned* dst = reinterpret_cast<unsigned*>(Ws + row * WS + cb);
      dst[0] = pk2bf(f0.x, f0.y); dst[1] = pk2bf(f0.z, f0.w);
      dst[2] = pk2bf(f1.x, f1.y); dst[3] = pk2bf(f1.z, f1.w);
    }
    if (k0 + 32 < D) {
      __builtin_prefetch(Ctx + (size_t)(m0 + t) * D + k0 + 32, 0, 1);
      __builtin_prefetch(Wo + (size_t)(n0 + (t >> 2)) * D + k0 + 32, 0, 1);
    }
#if defined(HAVE_ASYNC_LDS)
    wait_async0();
#endif
    __syncthreads();

    const v16bf a0 = make_afrag(Xs + (w * 32 + l) * XS, hi);
    const v16bf a1 = make_afrag(Xs + (w * 32 + 16 + l) * XS, hi);
#pragma unroll
    for (int nt = 0; nt < 4; ++nt) {
      const v16bf b = make_bfrag(Ws + (nt * 16 + l) * WS + hi * 16);
      acc[0][nt] = wmma_bf16(a0, b, acc[0][nt]);
      acc[1][nt] = wmma_bf16(a1, b, acc[1][nt]);
    }
  }
#pragma unroll
  for (int mt = 0; mt < 2; ++mt)
#pragma unroll
    for (int nt = 0; nt < 4; ++nt)
#pragma unroll
      for (int r = 0; r < 8; ++r) {
        const int m = m0 + w * 32 + mt * 16 + r + hi * 8;
        const int n = n0 + nt * 16 + l;
        Out[(size_t)m * 1024 + n] = acc[mt][nt][r];
      }
}

// ---------------------------------------------------------------------------
extern "C" void kernel_launch(void* const* d_in, const int* in_sizes, int n_in,
                              void* d_out, int out_size, void* d_ws, size_t ws_size,
                              hipStream_t stream) {
  (void)in_sizes; (void)n_in; (void)out_size; (void)ws_size;
  const float* X   = (const float*)d_in[0];
  const float* Wq  = (const float*)d_in[1];
  const float* Wk  = (const float*)d_in[2];
  const float* Wv  = (const float*)d_in[3];
  const float* Wo  = (const float*)d_in[4];
  const int*   pos = (const int*)d_in[5];

  const size_t MD = (size_t)4096 * 1024;          // one [B*S, D] bf16 matrix
  unsigned short* Qb  = (unsigned short*)d_ws;    // 8 MB each
  unsigned short* Kb  = Qb + MD;
  unsigned short* Vb  = Kb + MD;
  unsigned short* Ctx = Vb + MD;                  // total 32 MB of workspace
  float* Out = (float*)d_out;

  k_qkv_rope<<<dim3(16, 16, 3), 256, 0, stream>>>(X, Wq, Wk, Wv, pos, Qb, Kb, Vb);
  k_attn   <<<dim3(32, 32),     128, 0, stream>>>(Qb, Kb, Vb, Ctx);
  k_oproj  <<<dim3(16, 16),     256, 0, stream>>>(Ctx, Wo, Out);
}
